// GCN_LSTM_429496730135
// MI455X (gfx1250) — matrix operand order repaired and checked by reading, not compiled, not run on hardware
//
#include <hip/hip_runtime.h>
#include <hip/hip_bf16.h>

typedef __attribute__((ext_vector_type(16))) _Float16 v16h;
typedef __attribute__((ext_vector_type(8)))  float    v8f;

// ---------------------------------------------------------------------------
// CDNA5 async global->LDS staging via inline asm:
//   GLOBAL_LOAD_ASYNC_TO_LDS_B128 (tracked by ASYNCcnt, drained by
//   s_wait_asynccnt).  VDST = per-lane LDS byte address (low 32 bits of a
//   generic pointer into LDS), VADDR = 64-bit global address.
// ---------------------------------------------------------------------------
__device__ __forceinline__ void async_copy16(const float* gsrc, float* lds_dst) {
  unsigned lds_off = (unsigned)(size_t)lds_dst;              // LDS byte offset
  unsigned long long gaddr = (unsigned long long)(size_t)gsrc;
  asm volatile("global_load_async_to_lds_b128 %0, %1, off"
               :: "v"(lds_off), "v"(gaddr) : "memory");
}

__device__ __forceinline__ void wait_async0() {
  asm volatile("s_wait_asynccnt 0x0" ::: "memory");
}

// ---------------------------------------------------------------------------
// Graph-normalization prep kernels
// ---------------------------------------------------------------------------
__global__ void deg_init_kernel(float* deg, int N) {
  int i = blockIdx.x * blockDim.x + threadIdx.x;
  if (i < N) deg[i] = 1.0f;               // self loop contributes 1
}

__global__ void deg_scatter_kernel(const int* __restrict__ cols, float* deg, int E) {
  int e = blockIdx.x * blockDim.x + threadIdx.x;
  if (e < E) unsafeAtomicAdd(&deg[cols[e]], 1.0f);
}

__global__ void dinv_kernel(float* deg, int N) {
  int i = blockIdx.x * blockDim.x + threadIdx.x;
  if (i < N) { float d = deg[i]; deg[i] = d > 0.0f ? rsqrtf(d) : 0.0f; }
}

__global__ void norm_kernel(const int* __restrict__ rows, const int* __restrict__ cols,
                            const float* __restrict__ dinv, float* __restrict__ nrm, int E) {
  int e = blockIdx.x * blockDim.x + threadIdx.x;
  if (e < E) nrm[e] = dinv[rows[e]] * dinv[cols[e]];
}

// ---------------------------------------------------------------------------
// GCN dense stage:  T[N,64] = (relu?)(Hin[N,K]) @ W[K,64]   via WMMA f16->f32
// One block = 16 rows x 64 cols = 4 waves, each wave one 16x16 tile.
// A tile (16xK) staged raw to LDS via async-to-LDS B128 (ASYNCcnt), ReLU of
// the previous layer fused into the fragment build.
// ---------------------------------------------------------------------------
template<int K, bool RELU>
__global__ __launch_bounds__(128) void gcn_gemm_kernel(
    const float* __restrict__ Hin, const float* __restrict__ W, float* __restrict__ T)
{
  __shared__ float lds_a[16][K + 4];      // (K+4)*4B row stride is 16B-aligned
  const int tid = threadIdx.x;
  const int m0  = blockIdx.x * 16;

  {
    constexpr int QPR = K / 4;            // float4 chunks per row
    for (int idx = tid; idx < 16 * QPR; idx += 128) {
      int m = idx / QPR, q = idx - m * QPR;
      async_copy16(Hin + (size_t)(m0 + m) * K + 4 * q, &lds_a[m][4 * q]);
    }
    wait_async0();
  }
  __syncthreads();

  const int wave = tid >> 5, lane = tid & 31;
  const int n0 = wave * 16;
  const int lm = lane & 15;            // row (A) / col (B/C) within tile
  const int lh = lane >> 4;            // half-wave select

  auto lda = [&](int m, int k) -> float {
    float v = lds_a[m][k];
    return RELU ? (v > 0.0f ? v : 0.0f) : v;
  };

  v8f c = {};
#pragma unroll
  for (int k0 = 0; k0 < K; k0 += 32) {
    v16h a, b;
    // A fragment: 16-bit A 16x32 layout (ISA 7.12.2)
#pragma unroll
    for (int v = 0; v < 4; ++v) {
      int ka = k0 + lh * 8 + 2 * v;          // K 0..15 block
      a[2 * v]     = (_Float16)lda(lm, ka);
      a[2 * v + 1] = (_Float16)lda(lm, ka + 1);
      int kb = k0 + 16 + lh * 8 + 2 * v;     // K 16..31 block
      a[8 + 2 * v]     = (_Float16)lda(lm, kb);
      a[8 + 2 * v + 1] = (_Float16)lda(lm, kb + 1);
    }
    // B fragment: 32x16, lane = column, lanes 0-15 K=0..7,16..23, 16-31 K=8..15,24..31
#pragma unroll
    for (int v = 0; v < 8; ++v) {
      int kc = k0 + (v >> 2) * 16 + lh * 8 + 2 * (v & 3);
      b[2 * v]     = (_Float16)W[(size_t)kc * 64 + n0 + lm];
      b[2 * v + 1] = (_Float16)W[(size_t)(kc + 1) * 64 + n0 + lm];
    }
    c = __builtin_amdgcn_wmma_f32_16x16x32_f16(false, a, false, b, (short)0, c,
                                               false, false);
  }

  // C layout: VGPR r -> M = r (lanes 0-15), M = r+8 (lanes 16-31); N = lane%16
#pragma unroll
  for (int r = 0; r < 8; ++r) {
    int m = m0 + r + lh * 8;
    T[(size_t)m * 64 + n0 + lm] = c[r];
  }
}

// ---------------------------------------------------------------------------
// Aggregation: Out[i] = T[i]*dinv[i]^2 + bias   (self loop + bias init)
// then edge scatter adds T[row]*norm into Out[col] (atomic f32)
// ---------------------------------------------------------------------------
__global__ void agg_init_kernel(const float* __restrict__ T, const float* __restrict__ dinv,
                                const float* __restrict__ bias, float* __restrict__ Out, int N) {
  int idx = blockIdx.x * blockDim.x + threadIdx.x;
  if (idx >= N * 64) return;
  int i = idx >> 6, d = idx & 63;
  float di = dinv[i];
  Out[idx] = T[idx] * di * di + bias[d];
}

__global__ void edge_scatter_kernel(const float* __restrict__ T, const float* __restrict__ nrm,
                                    const int* __restrict__ rows, const int* __restrict__ cols,
                                    float* __restrict__ Out, int E) {
  int idx = blockIdx.x * blockDim.x + threadIdx.x;
  int e = idx >> 4;
  if (e >= E) return;
  int d0 = (idx & 15) << 2;
  int r = rows[e], cl = cols[e];
  float nm = nrm[e];
  const float4 tv = *(const float4*)(T + (size_t)r * 64 + d0);
  float* outp = Out + (size_t)cl * 64 + d0;
  unsafeAtomicAdd(outp + 0, tv.x * nm);
  unsafeAtomicAdd(outp + 1, tv.y * nm);
  unsafeAtomicAdd(outp + 2, tv.z * nm);
  unsafeAtomicAdd(outp + 3, tv.w * nm);
}

// ---------------------------------------------------------------------------
// Fused single-step LSTM (h0=c0=0):
//   gates[16,512] = relu(H[16,64]) @ W_ih^T   (32 waves of WMMA, K=64)
//   h1 = sigmoid(o) * tanh( sigmoid(i) * tanh(g) )
// One block handles 16 nodes; gates live in LDS, never hit global memory.
// ---------------------------------------------------------------------------
__global__ __launch_bounds__(1024) void lstm_fused_kernel(
    const float* __restrict__ Hin,   // [N,64] pre-ReLU conv5 aggregation
    const float* __restrict__ W_ih,  // [512,64]
    const float* __restrict__ b_ih, const float* __restrict__ b_hh,
    float* __restrict__ H1)          // [N,128]
{
  __shared__ float lds_h[16][68];    // 272B row stride, 16B-aligned
  __shared__ float lds_g[16][516];
  const int tid = threadIdx.x;
  const int m0  = blockIdx.x * 16;

  // stage node features raw (ReLU applied at fragment build)
  if (tid < 256) {                    // 16 rows x 16 float4
    int m = tid >> 4, q = tid & 15;
    async_copy16(Hin + (size_t)(m0 + m) * 64 + 4 * q, &lds_h[m][4 * q]);
  }
  wait_async0();
  __syncthreads();

  const int wave = tid >> 5, lane = tid & 31;
  const int n0 = wave * 16;            // gate-column tile (0..496)
  const int lm = lane & 15;
  const int lh = lane >> 4;

  auto ldh = [&](int m, int k) -> float {
    float v = lds_h[m][k];
    return v > 0.0f ? v : 0.0f;        // conv5 ReLU fused here
  };

  v8f c = {};
#pragma unroll
  for (int k0 = 0; k0 < 64; k0 += 32) {
    v16h a, b;
#pragma unroll
    for (int v = 0; v < 4; ++v) {
      int ka = k0 + lh * 8 + 2 * v;
      a[2 * v]     = (_Float16)ldh(lm, ka);
      a[2 * v + 1] = (_Float16)ldh(lm, ka + 1);
      int kb = k0 + 16 + lh * 8 + 2 * v;
      a[8 + 2 * v]     = (_Float16)ldh(lm, kb);
      a[8 + 2 * v + 1] = (_Float16)ldh(lm, kb + 1);
    }
    // B = W_ih^T, so B[k][n] = W_ih[n][k]; per-lane row of W_ih is contiguous
#pragma unroll
    for (int v = 0; v < 8; ++v) {
      int kc = k0 + (v >> 2) * 16 + lh * 8 + 2 * (v & 3);
      const float* wrow = W_ih + (size_t)(n0 + lm) * 64;
      b[2 * v]     = (_Float16)wrow[kc];
      b[2 * v + 1] = (_Float16)wrow[kc + 1];
    }
    c = __builtin_amdgcn_wmma_f32_16x16x32_f16(false, a, false, b, (short)0, c,
                                               false, false);
  }

  float bias = b_ih[n0 + lm] + b_hh[n0 + lm];
#pragma unroll
  for (int r = 0; r < 8; ++r) {
    lds_g[r + lh * 8][n0 + lm] = c[r] + bias;
  }
  __syncthreads();

  // pointwise gate math; gate order i,f,g,o ; f unused since c0 = 0
  for (int idx = tid; idx < 16 * 128; idx += 1024) {
    int m = idx >> 7, j = idx & 127;
    float gi = lds_g[m][j];
    float gg = lds_g[m][256 + j];
    float go = lds_g[m][384 + j];
    float iv = 1.0f / (1.0f + __expf(-gi));
    float gv = tanhf(gg);
    float ov = 1.0f / (1.0f + __expf(-go));
    float c1 = iv * gv;
    H1[(size_t)(m0 + m) * 128 + j] = ov * tanhf(c1);
  }
}

// ---------------------------------------------------------------------------
// FC head: out[n,c] = b_fc[c] + dot(h1[n,:], W_fc[c,:])   (K=128, C=10)
// ---------------------------------------------------------------------------
__global__ void fc_kernel(const float* __restrict__ H1, const float* __restrict__ W_fc,
                          const float* __restrict__ b_fc, float* __restrict__ out, int N) {
  int idx = blockIdx.x * blockDim.x + threadIdx.x;
  if (idx >= N * 10) return;
  int i = idx / 10, c = idx % 10;
  const float* h = H1 + (size_t)i * 128;
  const float* w = W_fc + (size_t)c * 128;
  float s = b_fc[c];
#pragma unroll
  for (int j = 0; j < 128; j += 4) {
    float4 hv = *(const float4*)(h + j);
    float4 wv = *(const float4*)(w + j);
    s += hv.x * wv.x + hv.y * wv.y + hv.z * wv.z + hv.w * wv.w;
  }
  out[idx] = s;
}

// ---------------------------------------------------------------------------
// Host launcher
// ---------------------------------------------------------------------------
extern "C" void kernel_launch(void* const* d_in, const int* in_sizes, int n_in,
                              void* d_out, int out_size, void* d_ws, size_t ws_size,
                              hipStream_t stream) {
  const float* X    = (const float*)d_in[0];
  const int*   ei   = (const int*)d_in[1];
  const float* W1   = (const float*)d_in[2];  const float* b1 = (const float*)d_in[3];
  const float* W2   = (const float*)d_in[4];  const float* b2 = (const float*)d_in[5];
  const float* W3   = (const float*)d_in[6];  const float* b3 = (const float*)d_in[7];
  const float* W4   = (const float*)d_in[8];  const float* b4 = (const float*)d_in[9];
  const float* W5   = (const float*)d_in[10]; const float* b5 = (const float*)d_in[11];
  const float* W_ih = (const float*)d_in[12];
  /* W_hh (index 13) unused: h0 = 0 */
  const float* b_ih = (const float*)d_in[14];
  const float* b_hh = (const float*)d_in[15];
  const float* W_fc = (const float*)d_in[16];
  const float* b_fc = (const float*)d_in[17];
  float* out = (float*)d_out;

  const int N = in_sizes[0] / 128;   // 50000
  const int E = in_sizes[1] / 2;     // 800000
  const int* rows = ei;
  const int* cols = ei + E;

  // workspace layout
  float* deg = (float*)d_ws;                 // N  (becomes dinv)
  float* nrm = deg + N;                      // E
  float* A   = nrm + E;                      // N*64
  float* B   = A + (size_t)N * 64;           // N*64
  float* C   = B + (size_t)N * 64;           // N*128 (GEMM staging + h1)

  const int T256 = 256;
  const int nb = (N + 15) / 16;              // 3125 tile rows (N % 16 == 0)

  // --- normalization ---
  deg_init_kernel<<<(N + T256 - 1) / T256, T256, 0, stream>>>(deg, N);
  deg_scatter_kernel<<<(E + T256 - 1) / T256, T256, 0, stream>>>(cols, deg, E);
  dinv_kernel<<<(N + T256 - 1) / T256, T256, 0, stream>>>(deg, N);
  norm_kernel<<<(E + T256 - 1) / T256, T256, 0, stream>>>(rows, cols, deg, nrm, E);

  const int gAgg = (N * 64 + T256 - 1) / T256;
  const int gSct = (E * 16 + T256 - 1) / T256;

  // --- conv1: X[N,128] -> A ---
  gcn_gemm_kernel<128, false><<<nb, 128, 0, stream>>>(X, W1, C);
  agg_init_kernel<<<gAgg, T256, 0, stream>>>(C, deg, b1, A, N);
  edge_scatter_kernel<<<gSct, T256, 0, stream>>>(C, nrm, rows, cols, A, E);
  // --- conv2: A -> B ---
  gcn_gemm_kernel<64, true><<<nb, 128, 0, stream>>>(A, W2, C);
  agg_init_kernel<<<gAgg, T256, 0, stream>>>(C, deg, b2, B, N);
  edge_scatter_kernel<<<gSct, T256, 0, stream>>>(C, nrm, rows, cols, B, E);
  // --- conv3: B -> A ---
  gcn_gemm_kernel<64, true><<<nb, 128, 0, stream>>>(B, W3, C);
  agg_init_kernel<<<gAgg, T256, 0, stream>>>(C, deg, b3, A, N);
  edge_scatter_kernel<<<gSct, T256, 0, stream>>>(C, nrm, rows, cols, A, E);
  // --- conv4: A -> B ---
  gcn_gemm_kernel<64, true><<<nb, 128, 0, stream>>>(A, W4, C);
  agg_init_kernel<<<gAgg, T256, 0, stream>>>(C, deg, b4, B, N);
  edge_scatter_kernel<<<gSct, T256, 0, stream>>>(C, nrm, rows, cols, B, E);
  // --- conv5: B -> A ---
  gcn_gemm_kernel<64, true><<<nb, 128, 0, stream>>>(B, W5, C);
  agg_init_kernel<<<gAgg, T256, 0, stream>>>(C, deg, b5, A, N);
  edge_scatter_kernel<<<gSct, T256, 0, stream>>>(C, nrm, rows, cols, A, E);

  // --- fused LSTM (relu(A) -> h1 in C) ---
  lstm_fused_kernel<<<nb, 1024, 0, stream>>>(A, W_ih, b_ih, b_hh, C);

  // --- FC head ---
  fc_kernel<<<(N * 10 + T256 - 1) / T256, T256, 0, stream>>>(C, W_fc, b_fc, out, N);
}
